// ConvLayer_8546984919120
// MI455X (gfx1250) — matrix operand order repaired and checked by reading, not compiled, not run on hardware
//
#include <hip/hip_runtime.h>
#include <hip/hip_bf16.h>

// ---- problem constants ----
constexpr int HH = 40, WW = 40;        // image H, W  (C = 1)
constexpr int KS = 5;                  // conv window
constexpr int HO = 36, WO = 36;        // output spatial
constexpr int PP = HO * WO;            // patches per image = 1296
constexpr int LL = KS * KS;            // patch length = 25
constexpr int MM = 384;                // inducing points
constexpr int NB = 64;                 // batch
constexpr int NPTOT = NB * PP;         // 82944 total patch columns
constexpr float JIT = 1e-6f;

constexpr int KUF_STRIDE = 392;        // bf16 elems per LDS row (384 + 8 pad)
constexpr int COLS_PER_WAVE = 32;      // two 16-col WMMA groups per wave
constexpr int WAVES = 4;               // waves per block
constexpr int BLOCK_COLS = COLS_PER_WAVE * WAVES;  // 128 columns per block
constexpr int NKT = MM / 32;           // 12 k-steps per 16-row block
constexpr int NMT = MM / 16;           // 24 row blocks

// LDS layout (bytes)
constexpr int ZS_OFF = 0;                               // bf16 [384][26]
constexpr int ZS_BYTES = MM * 26 * 2;                   // 19,968
constexpr int Z2_OFF = ZS_OFF + ZS_BYTES;               // f32 [384]
constexpr int Z2_BYTES = MM * 4;                        // 1,536
constexpr int PATCH_OFF = Z2_OFF + Z2_BYTES;            // f32 [128][26]
constexpr int PATCH_BYTES = BLOCK_COLS * 26 * 4;        // 13,312
constexpr int KUF_OFF = PATCH_OFF + PATCH_BYTES;        // bf16 [128][392]
constexpr int KUF_BYTES = BLOCK_COLS * KUF_STRIDE * 2;  // 100,352
constexpr int QS_OFF = KUF_OFF + KUF_BYTES;             // bf16 [2][16][392]
constexpr int QS_BYTES = 16 * KUF_STRIDE * 2;           // 12,544 per buffer
constexpr int SMEM_BYTES = QS_OFF + 2 * QS_BYTES;       // 160,256

typedef __attribute__((ext_vector_type(16))) __bf16 v16bf;
typedef __attribute__((ext_vector_type(8)))  float  v8f;
typedef __attribute__((ext_vector_type(4)))  unsigned int v4u;
typedef __attribute__((ext_vector_type(8)))  int v8i;
typedef __attribute__((ext_vector_type(4)))  int v4i;

// ------------------------------------------------------------------
// TDM: DMA one 16x384 bf16 row-block of Q into LDS, iterate mode so
// each 384-elem row lands at a 392-elem (padded) LDS stride.
// Issued once per calling wave; tracked with TENSORcnt.
// clang-23 toolchain: 6-arg builtin (extra int32x8 group, zeroed).
// ------------------------------------------------------------------
__device__ __forceinline__ void tdm_load_qblock(const __hip_bfloat16* gsrc,
                                                unsigned lds_off) {
  unsigned long long ga = (unsigned long long)(uintptr_t)gsrc;
  v4u g0;
  g0[0] = 1u;                                  // count=1 valid descriptor
  g0[1] = lds_off;                             // LDS byte address
  g0[2] = (unsigned)(ga & 0xFFFFFFFFu);        // global_addr[31:0]
  g0[3] = (unsigned)((ga >> 32) & 0x1FFFFFFu)  // global_addr[56:32]
          | (2u << 30);                        // type = 2 ("image")
  v8i g1;
  g1[0] = (1 << 16) | (1 << 19);  // data_size=2B, iterate_enable=1
  g1[1] = (int)(384u << 16);      // tensor_dim0[15:0]=384 (bits 79:48)
  g1[2] = (int)(384u << 16);      // dim0 hi=0 | tensor_dim1[15:0]=384
  g1[3] = (int)(384u << 16);      // dim1 hi=0 | tile_dim0=384
  g1[4] = 1;                      // tile_dim1=1 (one row per iteration)
  g1[5] = 384;                    // tensor_dim0_stride low
  g1[6] = (int)(384u << 16);      // stride0 hi=0 | tensor_dim1_stride low16
  g1[7] = 0;                      // tensor_dim1_stride hi
  v4i g2;
  g2[0] = 1;                      // tensor_dim2
  g2[1] = 392;                    // lds_addr_increment (data_size units)
  g2[2] = 384;                    // global_addr_increment low (one row)
  g2[3] = (int)(15u << 16);       // iterate_count = 15 -> 16 iterations
  v4i g3;
  g3[0] = 0; g3[1] = 0; g3[2] = 0; g3[3] = 0;
  v8i g4;
  g4[0] = 0; g4[1] = 0; g4[2] = 0; g4[3] = 0;
  g4[4] = 0; g4[5] = 0; g4[6] = 0; g4[7] = 0;
  __builtin_amdgcn_tensor_load_to_lds(g0, g1, g2, g3, g4, 0);
}

// ------------------------------------------------------------------
// Kuu = rbf(Z,Z) + jitter*I   (written into the Cholesky buffer)
// ------------------------------------------------------------------
__global__ __launch_bounds__(256) void k_kuu(const float* __restrict__ Z,
                                             const float* __restrict__ var_p,
                                             const float* __restrict__ ls_p,
                                             float* __restrict__ LU) {
  int idx = blockIdx.x * blockDim.x + threadIdx.x;
  if (idx >= MM * MM) return;
  int i = idx / MM, j = idx % MM;
  float ls = ls_p[0];
  float inv2 = 0.5f / (ls * ls);
  float sq = 0.f;
  for (int l = 0; l < LL; ++l) {
    float d = Z[i * LL + l] - Z[j * LL + l];
    sq += d * d;
  }
  float v = var_p[0] * __expf(-sq * inv2);
  if (i == j) v += JIT;
  LU[idx] = v;
}

// ------------------------------------------------------------------
// In-place Cholesky (lower) of LU, one block of 384 threads
// ------------------------------------------------------------------
__global__ __launch_bounds__(384) void k_chol(float* __restrict__ A) {
  int tid = threadIdx.x;
  for (int k = 0; k < MM; ++k) {
    if (tid == 0) A[k * MM + k] = sqrtf(A[k * MM + k]);
    __syncthreads();
    float dk = A[k * MM + k];
    for (int i = k + 1 + tid; i < MM; i += blockDim.x) A[i * MM + k] /= dk;
    __syncthreads();
    for (int i = k + 1 + tid; i < MM; i += blockDim.x) {
      float lik = A[i * MM + k];
      for (int j = k + 1; j <= i; ++j) A[i * MM + j] -= lik * A[j * MM + k];
    }
    __syncthreads();
  }
}

// ------------------------------------------------------------------
// LI = inv(L) (lower triangular); thread j owns column j
// ------------------------------------------------------------------
__global__ __launch_bounds__(384) void k_trinv(const float* __restrict__ L,
                                               float* __restrict__ LI) {
  int j = threadIdx.x;
  if (j >= MM) return;
  for (int i = 0; i < j; ++i) LI[i * MM + j] = 0.f;
  for (int i = j; i < MM; ++i) {
    float s = (i == j) ? 1.f : 0.f;
    for (int k = j; k < i; ++k) s -= L[i * MM + k] * LI[k * MM + j];
    LI[i * MM + j] = s / L[i * MM + i];
  }
}

// ------------------------------------------------------------------
// B = LI^T * LI  == Kuu^{-1}
// ------------------------------------------------------------------
__global__ __launch_bounds__(256) void k_bmat(const float* __restrict__ LI,
                                              float* __restrict__ Bm) {
  int idx = blockIdx.x * blockDim.x + threadIdx.x;
  if (idx >= MM * MM) return;
  int i = idx / MM, j = idx % MM;
  int k0 = i > j ? i : j;
  float s = 0.f;
  for (int k = k0; k < MM; ++k) s += LI[k * MM + i] * LI[k * MM + j];
  Bm[idx] = s;
}

// ------------------------------------------------------------------
// C = B * tril(q_sqrt)
// ------------------------------------------------------------------
__global__ __launch_bounds__(256) void k_cmat(const float* __restrict__ Bm,
                                              const float* __restrict__ qs,
                                              float* __restrict__ Cm) {
  int idx = blockIdx.x * blockDim.x + threadIdx.x;
  if (idx >= MM * MM) return;
  int i = idx / MM, j = idx % MM;
  float s = 0.f;
  for (int k = j; k < MM; ++k) s += Bm[i * MM + k] * qs[k * MM + j];
  Cm[idx] = s;
}

// ------------------------------------------------------------------
// Q = C*C^T - B  (stored as bf16 for the WMMA consumer)
// ------------------------------------------------------------------
__global__ __launch_bounds__(256) void k_qw(const float* __restrict__ Cm,
                                            const float* __restrict__ Bm,
                                            __hip_bfloat16* __restrict__ Qbf) {
  int idx = blockIdx.x * blockDim.x + threadIdx.x;
  if (idx >= MM * MM) return;
  int i = idx / MM, j = idx % MM;
  float s = 0.f;
  for (int k = 0; k < MM; ++k) s += Cm[i * MM + k] * Cm[j * MM + k];
  Qbf[idx] = __float2bfloat16(s - Bm[idx]);
}

// ------------------------------------------------------------------
// W = B * q_mu
// ------------------------------------------------------------------
__global__ __launch_bounds__(384) void k_w(const float* __restrict__ Bm,
                                           const float* __restrict__ qmu,
                                           float* __restrict__ Wv) {
  int i = threadIdx.x;
  if (i >= MM) return;
  float s = 0.f;
  for (int k = 0; k < MM; ++k) s += Bm[i * MM + k] * qmu[k];
  Wv[i] = s;
}

// ------------------------------------------------------------------
// Fused main kernel, 128 threads = 4 waves, 128 columns per block.
//  - build 128 Kuf columns (bf16) in LDS (im2col + RBF fused)
//  - B-fragments pinned in VGPRs (192 regs/wave); A fragments from
//    a TDM double-buffered LDS stage of Q row-blocks
//  - diag = colwise <Kuf, Q*Kuf>, mean = W^T Kuf, fused epilogue
// ------------------------------------------------------------------
__global__ __launch_bounds__(WAVES * 32) void k_main(
    const float* __restrict__ X, const float* __restrict__ Z,
    const float* __restrict__ var_p, const float* __restrict__ ls_p,
    const __hip_bfloat16* __restrict__ Qbf, const float* __restrict__ Wv,
    float* __restrict__ out) {
  extern __shared__ char smem[];
  __hip_bfloat16* zs = (__hip_bfloat16*)(smem + ZS_OFF);    // [384][26]
  float* z2 = (float*)(smem + Z2_OFF);                      // [384]
  float* patch = (float*)(smem + PATCH_OFF);                // [128][26]
  __hip_bfloat16* kuf = (__hip_bfloat16*)(smem + KUF_OFF);  // [128][392]

  int tid = threadIdx.x;
  int lane = tid & 31;
  int wave = tid >> 5;

  float var = var_p[0];
  float ls = ls_p[0];
  float inv2 = 0.5f / (ls * ls);

  // ---- stage Z as bf16 (whole block cooperates) ----
  for (int idx = tid; idx < MM * LL; idx += BLOCK_COLS) {
    int m = idx / LL, l = idx % LL;
    zs[m * 26 + l] = __float2bfloat16(Z[idx]);
  }
  __syncthreads();
  for (int m = tid; m < MM; m += BLOCK_COLS) {
    float s = 0.f;
    for (int l = 0; l < LL; ++l) {
      float z = __bfloat162float(zs[m * 26 + l]);
      s += z * z;
    }
    z2[m] = s;
  }

  // ---- stage 128 patches, one column per thread (im2col on the fly) ----
  {
    int colg = blockIdx.x * BLOCK_COLS + tid;
    int n = colg / PP, p = colg % PP;
    int oh = p / WO, ow = p % WO;
    const float* xb = X + n * (HH * WW) + oh * WW + ow;
    float p2 = 0.f;
    for (int ki = 0; ki < KS; ++ki)
      for (int kj = 0; kj < KS; ++kj) {
        float v = xb[ki * WW + kj];
        patch[tid * 26 + ki * KS + kj] = v;
        p2 += v * v;
      }
    patch[tid * 26 + 25] = p2;
  }
  __syncthreads();

  // ---- build this wave's 32 Kuf columns (bf16) in LDS ----
  for (int c = 0; c < COLS_PER_WAVE; ++c) {
    int cl = wave * COLS_PER_WAVE + c;
    float p2 = patch[cl * 26 + 25];
    for (int m = lane; m < MM; m += 32) {
      float dot = 0.f;
      for (int l = 0; l < LL; ++l)
        dot += __bfloat162float(zs[m * 26 + l]) * patch[cl * 26 + l];
      float sq = fmaxf(z2[m] + p2 - 2.f * dot, 0.f);
      kuf[cl * KUF_STRIDE + m] = __float2bfloat16(var * __expf(-sq * inv2));
    }
  }

  // ---- kick off TDM for the first Q row-block while kuf settles ----
  if (wave == 0) tdm_load_qblock(Qbf, QS_OFF);
  __syncthreads();  // kuf visible to all lanes/waves

  int hi = lane >> 4;  // lane-half per ISA fragment layout
  int ln = lane & 15;
  int c0 = wave * COLS_PER_WAVE + ln;  // local col, group 0
  int c1 = c0 + 16;                    // local col, group 1
  union U16 { v16bf v; float4 f[2]; };

  // ---- pin all 24 B-fragments (2 col groups x 12 k-steps) in VGPRs ----
  U16 breg0[NKT], breg1[NKT];
#pragma unroll
  for (int kt = 0; kt < NKT; ++kt) {
    const float4* bp0 = (const float4*)(const void*)(kuf + c0 * KUF_STRIDE +
                                                     kt * 32 + hi * 8);
    breg0[kt].f[0] = bp0[0];
    breg0[kt].f[1] = bp0[2];
    const float4* bp1 = (const float4*)(const void*)(kuf + c1 * KUF_STRIDE +
                                                     kt * 32 + hi * 8);
    breg1[kt].f[0] = bp1[0];
    breg1[kt].f[1] = bp1[2];
  }

  float diagp0 = 0.f, diagp1 = 0.f;

  for (int mt = 0; mt < NMT; ++mt) {
    int b = mt & 1;
    __syncthreads();  // all waves done reading buffer b^1 (iteration mt-1)
    if (wave == 0) {
      if (mt + 1 < NMT) {
        tdm_load_qblock(Qbf + (mt + 1) * 16 * MM, QS_OFF + (b ^ 1) * QS_BYTES);
        __builtin_amdgcn_s_wait_tensorcnt(1);  // block mt resident
      } else {
        __builtin_amdgcn_s_wait_tensorcnt(0);
      }
    }
    __syncthreads();  // publish buffer b
    const __hip_bfloat16* qsb =
        (const __hip_bfloat16*)(smem + QS_OFF + b * QS_BYTES);

    v8f acc0 = {0.f, 0.f, 0.f, 0.f, 0.f, 0.f, 0.f, 0.f};
    v8f acc1 = {0.f, 0.f, 0.f, 0.f, 0.f, 0.f, 0.f, 0.f};
#pragma unroll
    for (int kt = 0; kt < NKT; ++kt) {
      int k0 = kt * 32;
      // A fragment (row ln of the staged block): lane<16 -> K {k0..k0+7,
      // k0+16..k0+23}, lane>=16 -> K {k0+8..k0+15, k0+24..k0+31}
      U16 a;
      const float4* ap =
          (const float4*)(const void*)(qsb + ln * KUF_STRIDE + k0 + hi * 8);
      a.f[0] = ap[0];
      a.f[1] = ap[2];  // +16 bf16 elements
      acc0 = __builtin_amdgcn_wmma_f32_16x16x32_bf16(
          false, a.v, false, breg0[kt].v, (short)0, acc0, false, false);
      acc1 = __builtin_amdgcn_wmma_f32_16x16x32_bf16(
          false, a.v, false, breg1[kt].v, (short)0, acc1, false, false);
    }
    // C/D layout: VGPR j -> row (mt*16 + 8*hi + j), col ln
#pragma unroll
    for (int j = 0; j < 8; ++j) {
      int r = mt * 16 + hi * 8 + j;
      diagp0 += acc0[j] * __bfloat162float(kuf[c0 * KUF_STRIDE + r]);
      diagp1 += acc1[j] * __bfloat162float(kuf[c1 * KUF_STRIDE + r]);
    }
  }
  diagp0 += __shfl_down(diagp0, 16, 32);  // combine lane halves (wave32)
  diagp1 += __shfl_down(diagp1, 16, 32);

  // ---- mean: each of the 32 lanes owns one of this wave's columns ----
  {
    int cl = wave * COLS_PER_WAVE + lane;
    float mean = 0.f;
    for (int m = 0; m < MM; ++m)
      mean += Wv[m] * __bfloat162float(kuf[cl * KUF_STRIDE + m]);
    out[blockIdx.x * BLOCK_COLS + cl] = mean;  // mean [N,P] flat
  }
  // ---- variance: lanes 0..15 hold the reduced diagonals ----
  if (lane < 16) {
    int colg0 = blockIdx.x * BLOCK_COLS + c0;
    out[NPTOT + colg0] = var + diagp0;
    out[NPTOT + colg0 + 16] = var + diagp1;
  }
}

// ------------------------------------------------------------------
extern "C" void kernel_launch(void* const* d_in, const int* in_sizes, int n_in,
                              void* d_out, int out_size, void* d_ws,
                              size_t ws_size, hipStream_t stream) {
  const float* X = (const float*)d_in[0];
  const float* Z = (const float*)d_in[1];
  const float* q_mu = (const float*)d_in[2];
  const float* q_sqrt = (const float*)d_in[3];
  const float* variance = (const float*)d_in[4];
  const float* lengthscale = (const float*)d_in[5];
  float* out = (float*)d_out;

  char* ws = (char*)d_ws;
  constexpr size_t MAT = (size_t)MM * MM * sizeof(float);  // 589,824 B
  float* LU = (float*)(ws);
  float* LI = (float*)(ws + MAT);
  float* Bm = (float*)(ws + 2 * MAT);
  float* Cm = (float*)(ws + 3 * MAT);
  __hip_bfloat16* Qbf = (__hip_bfloat16*)(ws + 4 * MAT);
  float* Wv = (float*)(ws + 4 * MAT + (size_t)MM * MM * 2);
  // total ws use: 4*589824 + 294912 + 1536 = 2,655,744 bytes

  int mm_blocks = (MM * MM + 255) / 256;
  k_kuu<<<mm_blocks, 256, 0, stream>>>(Z, variance, lengthscale, LU);
  k_chol<<<1, 384, 0, stream>>>(LU);
  k_trinv<<<1, 384, 0, stream>>>(LU, LI);
  k_bmat<<<mm_blocks, 256, 0, stream>>>(LI, Bm);
  k_cmat<<<mm_blocks, 256, 0, stream>>>(Bm, q_sqrt, Cm);
  k_qw<<<mm_blocks, 256, 0, stream>>>(Cm, Bm, Qbf);
  k_w<<<1, 384, 0, stream>>>(Bm, q_mu, Wv);

  int nblocks = NPTOT / BLOCK_COLS;  // 648 blocks, 128 threads each
  k_main<<<nblocks, WAVES * 32, SMEM_BYTES, stream>>>(
      X, Z, variance, lengthscale, Qbf, Wv, out);
}
